// ifa_simfpn_66391604462434
// MI455X (gfx1250) — matrix operand order repaired and checked by reading, not compiled, not run on hardware
//
#include <hip/hip_runtime.h>
#include <hip/hip_bf16.h>

typedef __attribute__((ext_vector_type(16))) __bf16 v16bf;
typedef __attribute__((ext_vector_type(8)))  float  v8f;

#define BATCH  4
#define CFEAT  256
#define HOUT   128
#define LPIX   (HOUT*HOUT)     // 16384
#define BL_    (BATCH*LPIX)    // 65536 rows
#define MTILES (BL_/16)        // 4096
#define K1P    1216            // 1192 padded to 38*32
#define KS1    (K1P/32)        // 38

// ---- fragment index helpers (ISA 7.12.2 16-bit A 16x32 and B 32x16 layouts) ----
// A tile (16 rows x 32 K) = 512 halfs: lane = m + 16*bit3(kk), half = (kk&7) | (bit4(kk)<<3)
__device__ __forceinline__ int fragA_idx(int m, int k) {
  int kk = k & 31, ks = k >> 5;
  int lane = (m & 15) + (((kk >> 3) & 1) << 4);
  int half = (kk & 7) | ((kk >> 4) << 3);
  return ks * 512 + lane * 16 + half;
}
// B tile (32 K x 16 cols): lane = n + 16*bit3(kk), half = (kk&7) | (bit4(kk)<<3)
__device__ __forceinline__ int fragB_inner(int n, int kk) {
  int lane = (n & 15) + (((kk >> 3) & 1) << 4);
  return lane * 16 + ((kk & 7) | ((kk >> 4) << 3));
}

// ---------------- context build: gather + positional encoding -> bf16 A fragments ----------------
__global__ __launch_bounds__(256) void build_ctx(
    const float* __restrict__ x1, const float* __restrict__ x2,
    const float* __restrict__ x3, const float* __restrict__ x4,
    __bf16* __restrict__ Afrag)
{
  __shared__ unsigned int ldsU[KS1 * 256];          // 38 ksteps * 512 halfs = 38.9 KB
  __bf16* lds = reinterpret_cast<__bf16*>(ldsU);
  const int tid = threadIdx.x;
  const int m = tid & 15, g = tid >> 4;
  const int mtile = blockIdx.x;
  const int b = mtile >> 10;                        // 1024 m-tiles per batch image
  const int p = ((mtile & 1023) << 4) + m;
  const int py = p >> 7, px = p & 127;

  for (int i = tid; i < KS1 * 256; i += 256) ldsU[i] = 0u;   // zero incl. K padding
  __syncthreads();

  const float* xs[4] = {x1, x2, x3, x4};
  const int hhA[4] = {128, 64, 32, 16};

  const float gy = -1.f + (1.f/HOUT) + (2.f/HOUT) * (float)py;
  const float gx = -1.f + (1.f/HOUT) + (2.f/HOUT) * (float)px;

  #pragma unroll
  for (int lv = 0; lv < 4; ++lv) {
    const int hh = hhA[lv];
    const float fh = (float)hh;
    int iy = (int)rintf((gy + 1.f) * 0.5f * fh - 0.5f);
    iy = iy < 0 ? 0 : (iy > hh - 1 ? hh - 1 : iy);
    int ix = (int)rintf((gx + 1.f) * 0.5f * fh - 0.5f);
    ix = ix < 0 ? 0 : (ix > hh - 1 ? hh - 1 : ix);
    const float qy = -1.f + 1.f/fh + (2.f/fh) * (float)iy;
    const float qx = -1.f + 1.f/fh + (2.f/fh) * (float)ix;
    const float ry = (gy - qy) * fh;
    const float rx = (gx - qx) * fh;
    const int k0 = lv * 298;
    if (g == 0) {
      lds[fragA_idx(m, k0 + 0)] = (__bf16)ry;
      lds[fragA_idx(m, k0 + 1)] = (__bf16)rx;
    }
    for (int jj = g; jj < 20; jj += 16) {           // 20 freqs: rel_y*m[0..9], rel_x*m[0..9]
      const float f = exp2f((6.f / 9.f) * (float)(jj % 10));
      const float y = (jj < 10 ? ry : rx) * f;
      float s, c;
      __sincosf(y, &s, &c);
      lds[fragA_idx(m, k0 + 2  + jj)] = (__bf16)s;
      lds[fragA_idx(m, k0 + 22 + jj)] = (__bf16)c;
    }
    const float* xb = xs[lv] + (size_t)b * CFEAT * hh * hh;
    const int pix = iy * hh + ix;
    for (int c = g; c < CFEAT; c += 16) {
      float v = xb[(size_t)c * hh * hh + pix];
      lds[fragA_idx(m, k0 + 42 + c)] = (__bf16)v;
    }
  }
  __syncthreads();
  unsigned int* dst = reinterpret_cast<unsigned int*>(Afrag + (size_t)mtile * (KS1 * 512));
  for (int i = tid; i < KS1 * 256; i += 256) dst[i] = ldsU[i];
}

// ---------------- weight repack f32 [N,K] -> bf16 B fragments [ntile][kstep][512] ----------------
__global__ void pack_weights(const float* __restrict__ w, __bf16* __restrict__ dst,
                             int N, int K, int Npad, int Kp)
{
  int idx = blockIdx.x * blockDim.x + threadIdx.x;
  if (idx >= Npad * Kp) return;
  int n = idx / Kp, k = idx - n * Kp;
  float v = (n < N && k < K) ? w[(size_t)n * K + k] : 0.f;
  int ntile = n >> 4, ks = k >> 5, kk = k & 31;
  dst[(size_t)(ntile * (Kp >> 5) + ks) * 512 + fragB_inner(n, kk)] = (__bf16)v;
}

__global__ void zero_f32(float* p, int n) {
  int i = blockIdx.x * blockDim.x + threadIdx.x;
  if (i < n) p[i] = 0.f;
}

// ---------------- GEMM: per wave 16M x (NT*16)N, epilogue computes BN sum/sumsq ----------------
template <int NT>
__global__ __launch_bounds__(256) void gemm_bf16(
    const __bf16* __restrict__ A, const __bf16* __restrict__ W,
    float* __restrict__ Y, float* __restrict__ stats, int Ksteps, int N)
{
  __shared__ float sSum[NT * 16];
  __shared__ float sSq[NT * 16];
  const int tid = threadIdx.x;
  const int lane = tid & 31;
  const int mtile = blockIdx.x * 8 + (tid >> 5);
  const int ntile0 = blockIdx.y * NT;

  if (tid < NT * 16) { sSum[tid] = 0.f; sSq[tid] = 0.f; }
  __syncthreads();

  v8f acc[NT] = {};
  const __bf16* aP = A + (size_t)mtile * (Ksteps * 512) + lane * 16;
  for (int ks = 0; ks < Ksteps; ++ks) {
    v16bf a = *reinterpret_cast<const v16bf*>(aP + (size_t)ks * 512);
    if (ks + 4 < Ksteps) __builtin_prefetch(aP + (size_t)(ks + 4) * 512, 0, 1);
    #pragma unroll
    for (int nt = 0; nt < NT; ++nt) {
      const __bf16* bP = W + ((size_t)(ntile0 + nt) * Ksteps + ks) * 512 + lane * 16;
      v16bf bm = *reinterpret_cast<const v16bf*>(bP);
      acc[nt] = __builtin_amdgcn_wmma_f32_16x16x32_bf16(
          false, a, false, bm, (short)0, acc[nt], false, false);
    }
  }

  const int col16 = lane & 15;
  const int rowoff = (lane >> 4) << 3;              // lanes 16-31 hold M=8..15
  #pragma unroll
  for (int nt = 0; nt < NT; ++nt) {
    float s = 0.f, q = 0.f;
    #pragma unroll
    for (int i = 0; i < 8; ++i) { float v = acc[nt][i]; s += v; q += v * v; }
    s += __shfl_xor(s, 16, 32);
    q += __shfl_xor(q, 16, 32);
    if (stats && lane < 16) {
      atomicAdd(&sSum[nt * 16 + col16], s);         // ds_add_f32 block reduction
      atomicAdd(&sSq[nt * 16 + col16], q);
    }
    const int col = (ntile0 + nt) * 16 + col16;
    #pragma unroll
    for (int i = 0; i < 8; ++i) {
      int row = mtile * 16 + rowoff + i;
      Y[(size_t)row * N + col] = acc[nt][i];
    }
  }
  __syncthreads();
  if (stats && tid < NT * 16) {
    int col = ntile0 * 16 + tid;
    atomicAdd(&stats[col], sSum[tid]);
    atomicAdd(&stats[N + col], sSq[tid]);
  }
}

// ---------------- BN(train stats) + ReLU + repack to bf16 A fragments ----------------
__global__ __launch_bounds__(256) void bn_relu_pack(
    const float* __restrict__ Y, const float* __restrict__ stats,
    const float* __restrict__ gamma, const float* __restrict__ beta,
    __bf16* __restrict__ Afrag, int N)
{
  __shared__ __bf16 lds[16 * 512];                  // up to 16 ksteps (N=512)
  const int Ksteps = N >> 5;
  const int tid = threadIdx.x;
  const int m = tid >> 4, c0 = tid & 15;
  const int mtile = blockIdx.x;
  const size_t rowBase = (size_t)(mtile * 16 + m) * N;
  const float invBL = 1.f / (float)BL_;
  for (int c = c0; c < N; c += 16) {
    float mean  = stats[c] * invBL;
    float var   = stats[N + c] * invBL - mean * mean;
    float scale = gamma[c] * rsqrtf(var + 1e-5f);
    float shift = beta[c] - mean * scale;
    float v = fmaxf(Y[rowBase + c] * scale + shift, 0.f);
    lds[fragA_idx(m, c)] = (__bf16)v;
  }
  __syncthreads();
  const unsigned int* src = reinterpret_cast<const unsigned int*>(lds);
  unsigned int* dst = reinterpret_cast<unsigned int*>(Afrag + (size_t)mtile * (Ksteps * 512));
  for (int i = tid; i < Ksteps * 256; i += 256) dst[i] = src[i];
}

// ---------------- final 256 -> 19 GEMM, bias add, scatter to [B,19,h,w] ----------------
__global__ __launch_bounds__(256) void gemm_final(
    const __bf16* __restrict__ A, const __bf16* __restrict__ W,
    const float* __restrict__ bias, float* __restrict__ out, int Ksteps)
{
  const int tid = threadIdx.x;
  const int lane = tid & 31;
  const int mtile = blockIdx.x * 8 + (tid >> 5);
  v8f acc[2] = {};
  const __bf16* aP = A + (size_t)mtile * (Ksteps * 512) + lane * 16;
  for (int ks = 0; ks < Ksteps; ++ks) {
    v16bf a = *reinterpret_cast<const v16bf*>(aP + (size_t)ks * 512);
    #pragma unroll
    for (int nt = 0; nt < 2; ++nt) {
      const __bf16* bP = W + ((size_t)nt * Ksteps + ks) * 512 + lane * 16;
      v16bf bm = *reinterpret_cast<const v16bf*>(bP);
      acc[nt] = __builtin_amdgcn_wmma_f32_16x16x32_bf16(
          false, a, false, bm, (short)0, acc[nt], false, false);
    }
  }
  const int col16 = lane & 15;
  const int rowoff = (lane >> 4) << 3;
  #pragma unroll
  for (int nt = 0; nt < 2; ++nt) {
    int n = nt * 16 + col16;
    if (n >= 19) continue;
    float bv = bias[n];
    #pragma unroll
    for (int i = 0; i < 8; ++i) {
      int row = mtile * 16 + rowoff + i;
      int b = row >> 14;                            // L = 16384
      int p = row & (LPIX - 1);
      out[(((size_t)b * 19 + n) << 14) + p] = acc[nt][i] + bv;
    }
  }
}

extern "C" void kernel_launch(void* const* d_in, const int* in_sizes, int n_in,
                              void* d_out, int out_size, void* d_ws, size_t ws_size,
                              hipStream_t stream) {
  (void)in_sizes; (void)n_in; (void)out_size; (void)ws_size;
  const float* x1  = (const float*)d_in[0];
  const float* x2  = (const float*)d_in[1];
  const float* x3  = (const float*)d_in[2];
  const float* x4  = (const float*)d_in[3];
  const float* w1  = (const float*)d_in[4];
  const float* g1  = (const float*)d_in[6];
  const float* be1 = (const float*)d_in[7];
  const float* w2  = (const float*)d_in[8];
  const float* g2  = (const float*)d_in[10];
  const float* be2 = (const float*)d_in[11];
  const float* w3  = (const float*)d_in[12];
  const float* g3  = (const float*)d_in[14];
  const float* be3 = (const float*)d_in[15];
  const float* w4  = (const float*)d_in[16];
  const float* b4  = (const float*)d_in[17];
  // b1/b2/b3 are cancelled exactly by the BatchNorm mean subtraction.

  char* ws = (char*)d_ws;
  constexpr size_t A_OFF   = 0;
  constexpr size_t A_BYTES = (size_t)MTILES * KS1 * 1024;        // 159.4 MB bf16 fragments
  constexpr size_t Y_OFF   = A_OFF + A_BYTES;
  constexpr size_t Y_BYTES = (size_t)BL_ * 512 * 4;              // 134.2 MB f32
  constexpr size_t W1_OFF  = Y_OFF + Y_BYTES;
  constexpr size_t W1_BYTES = (size_t)32 * KS1 * 1024;
  constexpr size_t W2_OFF  = W1_OFF + W1_BYTES;
  constexpr size_t W2_BYTES = (size_t)16 * 16 * 1024;
  constexpr size_t W3_OFF  = W2_OFF + W2_BYTES;
  constexpr size_t W3_BYTES = (size_t)16 * 8 * 1024;
  constexpr size_t W4_OFF  = W3_OFF + W3_BYTES;
  constexpr size_t W4_BYTES = (size_t)2 * 8 * 1024;
  constexpr size_t ST_OFF  = W4_OFF + W4_BYTES;

  __bf16* Afrag = (__bf16*)(ws + A_OFF);
  float*  Ybuf  = (float*)(ws + Y_OFF);
  __bf16* W1f   = (__bf16*)(ws + W1_OFF);
  __bf16* W2f   = (__bf16*)(ws + W2_OFF);
  __bf16* W3f   = (__bf16*)(ws + W3_OFF);
  __bf16* W4f   = (__bf16*)(ws + W4_OFF);
  float*  st    = (float*)(ws + ST_OFF);
  float*  st1 = st;                 // 512 sum + 512 sumsq
  float*  st2 = st + 1024;          // 256 + 256
  float*  st3 = st + 1536;          // 256 + 256

  zero_f32<<<8, 256, 0, stream>>>(st, 2048);
  pack_weights<<<(512 * K1P + 255) / 256, 256, 0, stream>>>(w1, W1f, 512, 1192, 512, K1P);
  pack_weights<<<(256 * 512 + 255) / 256, 256, 0, stream>>>(w2, W2f, 256, 512, 256, 512);
  pack_weights<<<(256 * 256 + 255) / 256, 256, 0, stream>>>(w3, W3f, 256, 256, 256, 256);
  pack_weights<<<(32 * 256 + 255) / 256, 256, 0, stream>>>(w4, W4f, 19, 256, 32, 256);

  build_ctx<<<MTILES, 256, 0, stream>>>(x1, x2, x3, x4, Afrag);

  gemm_bf16<8><<<dim3(MTILES / 8, 4), 256, 0, stream>>>(Afrag, W1f, Ybuf, st1, KS1, 512);
  bn_relu_pack<<<MTILES, 256, 0, stream>>>(Ybuf, st1, g1, be1, Afrag, 512);

  gemm_bf16<8><<<dim3(MTILES / 8, 2), 256, 0, stream>>>(Afrag, W2f, Ybuf, st2, 16, 256);
  bn_relu_pack<<<MTILES, 256, 0, stream>>>(Ybuf, st2, g2, be2, Afrag, 256);

  gemm_bf16<8><<<dim3(MTILES / 8, 2), 256, 0, stream>>>(Afrag, W3f, Ybuf, st3, 8, 256);
  bn_relu_pack<<<MTILES, 256, 0, stream>>>(Ybuf, st3, g3, be3, Afrag, 256);

  gemm_final<<<dim3(MTILES / 8, 1), 256, 0, stream>>>(Afrag, W4f, b4, (float*)d_out, 8);
}
